// GATModule_34677565948516
// MI455X (gfx1250) — matrix operand order repaired and checked by reading, not compiled, not run on hardware
//
#include <hip/hip_runtime.h>
#include <hip/hip_bf16.h>
#include <cstdint>

#define N_NODES 50000
#define N_EDGES 640000
#define E_TOT   (N_EDGES + N_NODES)   // self-loops appended, PyG add_self_loops=True
#define D_IN    128
#define HID     64
#define HEADS   3
#define SLOPE   0.2f
#define GAT_EPS 1e-16f

typedef __attribute__((ext_vector_type(16))) _Float16 v16h;
typedef __attribute__((ext_vector_type(8)))  _Float16 v8h;
typedef __attribute__((ext_vector_type(8)))  float    v8f;

// ---------------------------------------------------------------- utilities

__device__ __forceinline__ float atomicMaxF(float* addr, float val) {
  unsigned int* ua = (unsigned int*)addr;
  unsigned int old = *(volatile unsigned int*)ua;
  while (__uint_as_float(old) < val) {
    unsigned int assumed = old;
    old = atomicCAS(ua, assumed, __float_as_uint(val));
    if (old == assumed) break;
  }
  return __uint_as_float(old);
}

__device__ __forceinline__ void edge_sd(const long long* __restrict__ ei,
                                        int e, int& s, int& d) {
  if (e < N_EDGES) {
    s = (int)ei[e];
    d = (int)ei[(size_t)N_EDGES + e];
  } else {
    s = d = e - N_EDGES;   // self loop
  }
}

// ---------------------------------------------------------------- convert (A operands)

__global__ void cvt_f32_to_f16(const float* __restrict__ in,
                               _Float16* __restrict__ out, int n) {
  int i = blockIdx.x * blockDim.x + threadIdx.x;
  if (i < n) out[i] = (_Float16)in[i];
}

// ---------------------------------------------------------------- B fragment packing
// Pack f32 B[K,N] into per-lane WMMA fragment order so the GEMM can load each
// 32x16 B fragment as ONE contiguous 32-byte load per lane:
//   Bp[((tn*(K/32) + kb)*32 + lane)*16 + i]  (halves)
//   col = tn*16 + (lane&15);  k = kb*32 + (lane>=16 ? 16 : 0) + i
__global__ void pack_b_f16(const float* __restrict__ B,
                           _Float16* __restrict__ Bp, int K, int N) {
  int t = blockIdx.x * blockDim.x + threadIdx.x;
  int total = K * N;
  if (t >= total) return;
  int i    = t & 15;
  int lane = (t >> 4) & 31;
  int kb   = (t >> 9) % (K >> 5);
  int tn   = t / ((K >> 5) << 9);
  int col  = (tn << 4) + (lane & 15);
  int k    = (kb << 5) + ((lane >= 16) ? 16 : 0) + i;
  Bp[t] = (_Float16)B[(size_t)k * N + col];
}

// ---------------------------------------------------------------- WMMA GEMM
// C[M,N] = A[M,K] (f16 row-major) x B (pre-packed f16 fragments), f32 out.
// M%16==0, N%64==0, K%32==0 (all shapes here qualify).
// One wave computes a 16x64 strip: 4 accumulators, A fragment loaded once per
// k-step, 4 back-to-back v_wmma_f32_16x16x32_f16 per k-step.
__global__ void wmma_gemm_f16(const _Float16* __restrict__ A,
                              const _Float16* __restrict__ Bp,
                              float* __restrict__ C,
                              int M, int N, int K) {
  const int wave = threadIdx.x >> 5;
  const int lane = threadIdx.x & 31;
  const int nbt  = N >> 6;                       // 16x64 strips per row-tile
  const int wid  = blockIdx.x * (blockDim.x >> 5) + wave;
  const int total = (M >> 4) * nbt;
  if (wid >= total) return;                      // uniform per-wave: EXEC all-ones
  const int tm = wid / nbt;
  const int tb = wid % nbt;
  const int r0 = tm << 4;
  const int rl = lane & 15;
  const bool hi = lane >= 16;
  const int kblocks = K >> 5;

  v8f acc0 = {}, acc1 = {}, acc2 = {}, acc3 = {};

  const _Float16* arow_base = A + (size_t)(r0 + rl) * K + (hi ? 16 : 0);
  const _Float16* bbase = Bp + ((size_t)(tb << 2) * kblocks * 32 + lane) * 16;

  for (int kb = 0; kb < kblocks; ++kb) {
    // A fragment (16x32): lanes 0-15 hold K {+0..15},
    // lanes 16-31 hold K {+16..31} -> two contiguous 16B chunks.
    const _Float16* arow = arow_base + (kb << 5);
    v8h alo = *(const v8h*)(arow);
    v8h ahi = *(const v8h*)(arow + 8);
    v16h a = __builtin_shufflevector(alo, ahi, 0, 1, 2, 3, 4, 5, 6, 7,
                                     8, 9, 10, 11, 12, 13, 14, 15);

    // 4 packed B fragments: one 32B contiguous load each
    const size_t kstep = (size_t)kb * 32 * 16;
    const size_t tstep = (size_t)kblocks * 32 * 16;
    v16h b0 = *(const v16h*)(bbase + kstep);
    v16h b1 = *(const v16h*)(bbase + tstep + kstep);
    v16h b2 = *(const v16h*)(bbase + 2 * tstep + kstep);
    v16h b3 = *(const v16h*)(bbase + 3 * tstep + kstep);

    acc0 = __builtin_amdgcn_wmma_f32_16x16x32_f16(false, a, false, b0, (short)0, acc0, false, false);
    acc1 = __builtin_amdgcn_wmma_f32_16x16x32_f16(false, a, false, b1, (short)0, acc1, false, false);
    acc2 = __builtin_amdgcn_wmma_f32_16x16x32_f16(false, a, false, b2, (short)0, acc2, false, false);
    acc3 = __builtin_amdgcn_wmma_f32_16x16x32_f16(false, a, false, b3, (short)0, acc3, false, false);
  }

  // D layout: VGPR j -> row r0 + j (+8 for lanes 16-31), col = c0 + (lane&15)
  float* crow = C + (size_t)(r0 + (hi ? 8 : 0)) * N + (tb << 6) + rl;
#pragma unroll
  for (int j = 0; j < 8; ++j) {
    crow[(size_t)j * N]      = acc0[j];
    crow[(size_t)j * N + 16] = acc1[j];
    crow[(size_t)j * N + 32] = acc2[j];
    crow[(size_t)j * N + 48] = acc3[j];
  }
}

// ---------------------------------------------------------------- attention dots
// asrc[n,h] = sum_c xw[n,h,c]*a_src[h,c]; adst likewise. One thread per (n,h).
__global__ void alpha_kernel(const float* __restrict__ xw,
                             const float* __restrict__ a_src,
                             const float* __restrict__ a_dst,
                             float* __restrict__ asrc,
                             float* __restrict__ adst, int heads) {
  int idx = blockIdx.x * blockDim.x + threadIdx.x;    // n*heads + h
  if (idx >= N_NODES * heads) return;
  int h = idx % heads;
  const float* row = xw + (size_t)idx * HID;
  float s = 0.f, d = 0.f;
#pragma unroll 4
  for (int c = 0; c < HID; ++c) {
    float v = row[c];
    s += v * a_src[h * HID + c];
    d += v * a_dst[h * HID + c];
  }
  asrc[idx] = s;
  adst[idx] = d;
}

// ---------------------------------------------------------------- init
__global__ void init_kernel(float* __restrict__ m, float* __restrict__ denom,
                            int nh, float* __restrict__ agg, int aggn) {
  int i = blockIdx.x * blockDim.x + threadIdx.x;
  if (i < nh) { m[i] = -INFINITY; denom[i] = 0.f; }
  if (i < aggn) agg[i] = 0.f;
}

// ---------------------------------------------------------------- edge pass 1: logits + segment max
__global__ void edge_logits_max(const long long* __restrict__ ei,
                                const float* __restrict__ asrc,
                                const float* __restrict__ adst,
                                float* __restrict__ elog,
                                float* __restrict__ m, int heads) {
  int idx = blockIdx.x * blockDim.x + threadIdx.x;    // e*heads + h
  if (idx >= E_TOT * heads) return;
  int e = idx / heads, h = idx % heads;
  int s, d; edge_sd(ei, e, s, d);
  float x = asrc[s * heads + h] + adst[d * heads + h];
  x = x > 0.f ? x : SLOPE * x;                        // leaky_relu(0.2)
  elog[idx] = x;
  atomicMaxF(&m[d * heads + h], x);
}

// ---------------------------------------------------------------- edge pass 2: exp + segment sum
__global__ void edge_exp_sum(const long long* __restrict__ ei,
                             const float* __restrict__ m,
                             float* __restrict__ elog,
                             float* __restrict__ denom, int heads) {
  int idx = blockIdx.x * blockDim.x + threadIdx.x;
  if (idx >= E_TOT * heads) return;
  int e = idx / heads, h = idx % heads;
  int s, d; edge_sd(ei, e, s, d);
  (void)s;
  float ex = __expf(elog[idx] - m[d * heads + h]);
  elog[idx] = ex;
  atomicAdd(&denom[d * heads + h], ex);
}

// ---------------------------------------------------------------- edge pass 3: weighted scatter-add
// One thread per (edge, head, channel); consecutive threads = consecutive
// channels -> coalesced gathers of xw[src] and coalesced atomics on agg[dst].
__global__ void edge_aggregate(const long long* __restrict__ ei,
                               const float* __restrict__ elog,
                               const float* __restrict__ denom,
                               const float* __restrict__ xw,
                               float* __restrict__ agg, int heads) {
  long long t = (long long)blockIdx.x * blockDim.x + threadIdx.x;
  long long tot = (long long)E_TOT * heads * HID;
  if (t >= tot) return;
  int c = (int)(t % HID);
  int h = (int)((t / HID) % heads);
  int e = (int)(t / ((long long)HID * heads));
  int s, d; edge_sd(ei, e, s, d);
  float alpha = elog[(size_t)e * heads + h] / (denom[d * heads + h] + GAT_EPS);
  atomicAdd(&agg[((size_t)d * heads + h) * HID + c],
            xw[((size_t)s * heads + h) * HID + c] * alpha);
}

// ---------------------------------------------------------------- epilogues
__global__ void bias_relu_inplace(float* __restrict__ io,
                                  const float* __restrict__ bias,
                                  int n, int cols) {
  int i = blockIdx.x * blockDim.x + threadIdx.x;
  if (i >= n) return;
  float v = io[i] + bias[i % cols];
  io[i] = v > 0.f ? v : 0.f;
}

__global__ void bias_relu_out(const float* __restrict__ in,
                              const float* __restrict__ bias,
                              float* __restrict__ out, int n, int cols) {
  int i = blockIdx.x * blockDim.x + threadIdx.x;
  if (i >= n) return;
  float v = in[i] + bias[i % cols];
  out[i] = v > 0.f ? v : 0.f;
}

// ---------------------------------------------------------------- launch

static inline unsigned int gridFor(long long n, int b) {
  return (unsigned int)((n + b - 1) / b);
}

extern "C" void kernel_launch(void* const* d_in, const int* in_sizes, int n_in,
                              void* d_out, int out_size, void* d_ws, size_t ws_size,
                              hipStream_t stream) {
  (void)in_sizes; (void)n_in; (void)out_size; (void)ws_size;
  const float*     x     = (const float*)d_in[0];
  const long long* ei    = (const long long*)d_in[1];   // int64 [2, 640000]
  const float*     W1    = (const float*)d_in[2];       // [128,192]
  const float*     as1   = (const float*)d_in[3];       // [3,64]
  const float*     ad1   = (const float*)d_in[4];
  const float*     b1    = (const float*)d_in[5];       // [192]
  const float*     W2    = (const float*)d_in[6];       // [192,64]
  const float*     as2   = (const float*)d_in[7];       // [1,64]
  const float*     ad2   = (const float*)d_in[8];
  const float*     b2    = (const float*)d_in[9];       // [64]
  float* out = (float*)d_out;                            // [50000,64]

  // ---- workspace layout (with reuse) ----
  char* ws = (char*)d_ws;
  size_t off = 0;
  auto alloc = [&](size_t bytes) -> void* {
    void* p = ws + off;
    off += (bytes + 255) & ~(size_t)255;
    return p;
  };
  const size_t NF1 = (size_t)N_NODES * HEADS * HID;     // 9.6M floats
  const size_t NF2 = (size_t)N_NODES * HID;             // 3.2M floats

  float*    bufA  = (float*)   alloc(NF1 * 4);          // xw1, later agg2
  float*    bufB  = (float*)   alloc(NF1 * 4);          // agg1/h1, later xw2
  _Float16* bufC  = (_Float16*)alloc(NF1 * 2);          // Xh, later h1h
  _Float16* W1p   = (_Float16*)alloc((size_t)D_IN * HEADS * HID * 2);
  _Float16* W2p   = (_Float16*)alloc((size_t)HEADS * HID * HID * 2);
  float*    asrc1 = (float*)   alloc((size_t)N_NODES * HEADS * 4);
  float*    adst1 = (float*)   alloc((size_t)N_NODES * HEADS * 4);
  float*    m1    = (float*)   alloc((size_t)N_NODES * HEADS * 4);
  float*    den1  = (float*)   alloc((size_t)N_NODES * HEADS * 4);
  float*    asrc2 = (float*)   alloc((size_t)N_NODES * 4);
  float*    adst2 = (float*)   alloc((size_t)N_NODES * 4);
  float*    m2    = (float*)   alloc((size_t)N_NODES * 4);
  float*    den2  = (float*)   alloc((size_t)N_NODES * 4);
  float*    elog  = (float*)   alloc((size_t)E_TOT * HEADS * 4); // reused layer 2

  const int B = 256;

  // ======== layer 1 ========
  cvt_f32_to_f16<<<gridFor((long long)N_NODES * D_IN, B), B, 0, stream>>>(
      x, bufC, N_NODES * D_IN);
  pack_b_f16<<<gridFor((long long)D_IN * HEADS * HID, B), B, 0, stream>>>(
      W1, W1p, D_IN, HEADS * HID);

  // xw1 = Xh @ W1 : [50000,128]x[128,192]; 3125 row-tiles x 3 strips, 4 waves/blk
  {
    int strips = (N_NODES / 16) * ((HEADS * HID) / 64);
    wmma_gemm_f16<<<gridFor(strips, 4), 128, 0, stream>>>(
        bufC, W1p, bufA, N_NODES, HEADS * HID, D_IN);
  }

  alpha_kernel<<<gridFor((long long)N_NODES * HEADS, B), B, 0, stream>>>(
      bufA, as1, ad1, asrc1, adst1, HEADS);

  init_kernel<<<gridFor((long long)NF1, B), B, 0, stream>>>(
      m1, den1, N_NODES * HEADS, bufB, (int)NF1);

  edge_logits_max<<<gridFor((long long)E_TOT * HEADS, B), B, 0, stream>>>(
      ei, asrc1, adst1, elog, m1, HEADS);
  edge_exp_sum<<<gridFor((long long)E_TOT * HEADS, B), B, 0, stream>>>(
      ei, m1, elog, den1, HEADS);
  edge_aggregate<<<gridFor((long long)E_TOT * HEADS * HID, B), B, 0, stream>>>(
      ei, elog, den1, bufA, bufB, HEADS);

  bias_relu_inplace<<<gridFor((long long)NF1, B), B, 0, stream>>>(
      bufB, b1, (int)NF1, HEADS * HID);

  // ======== layer 2 ========
  cvt_f32_to_f16<<<gridFor((long long)NF1, B), B, 0, stream>>>(
      bufB, bufC, (int)NF1);                            // h1 -> f16
  pack_b_f16<<<gridFor((long long)HEADS * HID * HID, B), B, 0, stream>>>(
      W2, W2p, HEADS * HID, HID);

  // xw2 = h1 @ W2 : [50000,192]x[192,64] -> bufB (h1 f32 no longer needed)
  {
    int strips = (N_NODES / 16) * (HID / 64);
    wmma_gemm_f16<<<gridFor(strips, 4), 128, 0, stream>>>(
        bufC, W2p, bufB, N_NODES, HID, HEADS * HID);
  }

  alpha_kernel<<<gridFor((long long)N_NODES, B), B, 0, stream>>>(
      bufB, as2, ad2, asrc2, adst2, 1);

  init_kernel<<<gridFor((long long)NF2, B), B, 0, stream>>>(
      m2, den2, N_NODES, bufA, (int)NF2);               // agg2 in bufA

  edge_logits_max<<<gridFor((long long)E_TOT, B), B, 0, stream>>>(
      ei, asrc2, adst2, elog, m2, 1);
  edge_exp_sum<<<gridFor((long long)E_TOT, B), B, 0, stream>>>(
      ei, m2, elog, den2, 1);
  edge_aggregate<<<gridFor((long long)E_TOT * HID, B), B, 0, stream>>>(
      ei, elog, den2, bufB, bufA, 1);

  bias_relu_out<<<gridFor((long long)NF2, B), B, 0, stream>>>(
      bufA, b2, out, (int)NF2, HID);
}
